// MultiHybridNeuralNetInterposePufNet_75677323755967
// MI455X (gfx1250) — compile-verified, matrix operand order checked by reading, not run on hardware
//
#include <hip/hip_runtime.h>

typedef __attribute__((ext_vector_type(16))) _Float16 v16h;
typedef __attribute__((ext_vector_type(8)))  _Float16 v8h;
typedef __attribute__((ext_vector_type(8)))  float    v8f;
typedef __attribute__((ext_vector_type(4)))  float    v4f;

#define SS 64     // S
#define PP 4      // P
#define HH 256    // H
#define KK 8      // K arbiters
#define BT 128    // batch tile per workgroup
#define NT 8      // BT/16 N-tiles

#define W1_ELEMS (PP * HH * SS)   // 65536
#define W2_ELEMS (PP * HH * HH)   // 262144

// D = A*B + C, fp16 in / fp32 acc, 16x16x32
__device__ __forceinline__ v8f wmma16(v16h a, v16h b, v8f c) {
  return __builtin_amdgcn_wmma_f32_16x16x32_f16(false, a, false, b, (short)0, c,
                                                false, false);
}

// Fragment from f16 memory (LDS or global): two contiguous 8-half chunks
// at +0 and +16 halves (K {0..7} and K {16..23} relative to lane's K base).
template <typename PtrT>
__device__ __forceinline__ v16h ldFrag(PtrT p) {
  v8h lo = *(const v8h*)(p);
  v8h hi = *(const v8h*)(p + 16);
  v16h r;
#pragma unroll
  for (int i = 0; i < 8; ++i) { r[i] = lo[i]; r[i + 8] = hi[i]; }
  return r;
}

// ---------------- prep: one-time f32 -> f16 weight conversion ----------------
__global__ __launch_bounds__(256) void cvt_weights_kernel(
    const float* __restrict__ W1, const float* __restrict__ W2,
    _Float16* __restrict__ W1h, _Float16* __restrict__ W2h) {
  int i = blockIdx.x * 256 + threadIdx.x;
  if (i < W1_ELEMS) W1h[i] = (_Float16)W1[i];
  if (i < W2_ELEMS) W2h[i] = (_Float16)W2[i];
}

__global__ __launch_bounds__(256) void puf_fused_kernel(
    const float* __restrict__ x,
    const _Float16* __restrict__ W1h, const float* __restrict__ b1,
    const _Float16* __restrict__ W2h, const float* __restrict__ b2,
    const float* __restrict__ W3,  const float* __restrict__ b3,
    const float* __restrict__ Wy,
    float* __restrict__ out, int Bn) {
  __shared__ __align__(16) _Float16 fB[BT * SS];    // f features, [b][s] f16
  __shared__ __align__(16) _Float16 h1T[BT * HH];   // hidden1,   [b][h] f16
  __shared__ float rawAcc[BT];                      // W3 reduction target
  __shared__ float WyL[KK * (SS + 2)];              // staged arbiter weights

  const int tid   = threadIdx.x;
  const int wave  = tid >> 5;
  const int lane  = tid & 31;
  const int laneM = lane & 15;          // row (A) / col (B) within 16
  const int kSel  = lane >> 4;          // 0: K {0..7,16..23}, 1: K {8..15,24..31}
  const int p        = blockIdx.y;
  const int tileBase = blockIdx.x * BT;
  const int mbase0   = wave * 32;       // this wave owns rows [mbase0, mbase0+32)

  // ---------------- Phase 0: features f + staging ----------------
  for (int i = tid; i < KK * (SS + 2); i += 256)
    WyL[i] = Wy[p * KK * (SS + 2) + i];
  if (tid < BT) {
    rawAcc[tid] = b3[p];
    const float* xr = x + (size_t)(tileBase + tid) * SS;
    float t[SS];
#pragma unroll
    for (int i = 0; i < SS / 4; ++i) {
      v4f v = *(const v4f*)(xr + i * 4);
#pragma unroll
      for (int j = 0; j < 4; ++j) t[i * 4 + j] = 1.f - 2.f * v[j];
    }
#pragma unroll
    for (int i = SS - 2; i >= 0; --i) t[i] *= t[i + 1];  // reverse cumprod
#pragma unroll
    for (int c = 0; c < SS / 8; ++c) {
      v8h hv;
#pragma unroll
      for (int j = 0; j < 8; ++j) hv[j] = (_Float16)t[c * 8 + j];
      *(v8h*)(fB + tid * SS + c * 8) = hv;
    }
  }
  __syncthreads();

  // ---------------- GEMM1: h1 = relu(W1 * f + b1), [256 x BT] ----------------
  // B-fragments (f) loaded once per nt, reused across both M-tiles.
  {
    const _Float16* W1p = W1h + p * HH * SS;
    const float* b1p = b1 + p * HH;
    v16h a0[2], a1[2];
    v4f bb0[2], bb1[2];
#pragma unroll
    for (int mt = 0; mt < 2; ++mt) {
      const _Float16* arow = W1p + (mbase0 + mt * 16 + laneM) * SS + kSel * 8;
      a0[mt] = ldFrag(arow);            // k-step 0 (K 0..31)
      a1[mt] = ldFrag(arow + 32);       // k-step 1 (K 32..63)
      const int h0 = mbase0 + mt * 16 + kSel * 8;
      bb0[mt] = *(const v4f*)(b1p + h0);
      bb1[mt] = *(const v4f*)(b1p + h0 + 4);
    }
#pragma unroll
    for (int nt = 0; nt < NT; ++nt) {
      const _Float16* bcol = fB + (nt * 16 + laneM) * SS + kSel * 8;
      v16h bf0 = ldFrag(bcol);
      v16h bf1 = ldFrag(bcol + 32);
#pragma unroll
      for (int mt = 0; mt < 2; ++mt) {
        v8f acc = {0.f, 0.f, 0.f, 0.f, 0.f, 0.f, 0.f, 0.f};
        acc = wmma16(a0[mt], bf0, acc);
        acc = wmma16(a1[mt], bf1, acc);
        v8h hv;
#pragma unroll
        for (int r = 0; r < 4; ++r) {
          hv[r]     = (_Float16)fmaxf(acc[r]     + bb0[mt][r], 0.f);
          hv[r + 4] = (_Float16)fmaxf(acc[r + 4] + bb1[mt][r], 0.f);
        }
        *(v8h*)(h1T + (nt * 16 + laneM) * HH + mbase0 + mt * 16 + kSel * 8) = hv;
      }
    }
  }
  __syncthreads();

  // ------------- GEMM2: h2 = relu(W2 * h1 + b2), fused W3 reduction ----------
  // k-outer: each B-fragment (h1) loaded once, feeds both M-tile accumulators.
  {
    const _Float16* W2p = W2h + p * HH * HH;
    const float* b2p = b2 + p * HH;
    const float* W3p = W3 + p * HH;
    const _Float16* arow0 = W2p + (size_t)(mbase0 + laneM) * HH + kSel * 8;
    const _Float16* arow1 = W2p + (size_t)(mbase0 + 16 + laneM) * HH + kSel * 8;
    v8f acc[2][NT];
#pragma unroll
    for (int mt = 0; mt < 2; ++mt)
#pragma unroll
      for (int nt = 0; nt < NT; ++nt)
        acc[mt][nt] = (v8f){0.f, 0.f, 0.f, 0.f, 0.f, 0.f, 0.f, 0.f};
#pragma unroll
    for (int k = 0; k < 8; ++k) {                // K = 256 -> 8 k-steps
      v16h a0 = ldFrag(arow0 + k * 32);
      v16h a1 = ldFrag(arow1 + k * 32);
#pragma unroll
      for (int nt = 0; nt < NT; ++nt) {
        v16h bf = ldFrag(h1T + (nt * 16 + laneM) * HH + k * 32 + kSel * 8);
        acc[0][nt] = wmma16(a0, bf, acc[0][nt]);
        acc[1][nt] = wmma16(a1, bf, acc[1][nt]);
      }
    }
#pragma unroll
    for (int mt = 0; mt < 2; ++mt) {
      const int h0 = mbase0 + mt * 16 + kSel * 8;
      v4f c0 = *(const v4f*)(b2p + h0);
      v4f c1 = *(const v4f*)(b2p + h0 + 4);
      v4f w0 = *(const v4f*)(W3p + h0);
      v4f w1 = *(const v4f*)(W3p + h0 + 4);
#pragma unroll
      for (int nt = 0; nt < NT; ++nt) {
        float s = 0.f;
#pragma unroll
        for (int r = 0; r < 4; ++r) {
          s += fmaxf(acc[mt][nt][r]     + c0[r], 0.f) * w0[r];
          s += fmaxf(acc[mt][nt][r + 4] + c1[r], 0.f) * w1[r];
        }
        atomicAdd(&rawAcc[nt * 16 + laneM], s);   // ds_add_f32
      }
    }
  }
  __syncthreads();

  // ---------------- Arbiter: yf assembly folded algebraically ----------------
  if (tid < BT) {
    float xo = 1.f / (1.f + __expf(-rawAcc[tid]));
    float ax = fabsf(xo);
    float fv[SS];
#pragma unroll
    for (int c = 0; c < SS / 8; ++c) {           // 8x ds_load_b128
      v8h hv = *(const v8h*)(fB + tid * SS + c * 8);
#pragma unroll
      for (int j = 0; j < 8; ++j) fv[c * 8 + j] = (float)hv[j];
    }
    float prod = 1.f;
#pragma unroll
    for (int k = 0; k < KK; ++k) {
      const float* wy = WyL + k * (SS + 2);
      float s1 = wy[32] * fv[32];   // duplicated pivot term
      float s2 = wy[65];            // trailing axo term
#pragma unroll
      for (int j = 0; j < 32; ++j) {
        s1 += wy[j] * fv[j];
        s2 += wy[33 + j] * fv[32 + j];
      }
      prod *= xo * s1 + ax * s2;
    }
    out[(size_t)p * Bn + tileBase + tid] = 1.f / (1.f + __expf(-prod));
  }
}

extern "C" void kernel_launch(void* const* d_in, const int* in_sizes, int n_in,
                              void* d_out, int out_size, void* d_ws, size_t ws_size,
                              hipStream_t stream) {
  (void)n_in; (void)ws_size; (void)out_size;
  const float* x  = (const float*)d_in[0];
  const float* W1 = (const float*)d_in[1];
  const float* b1 = (const float*)d_in[2];
  const float* W2 = (const float*)d_in[3];
  const float* b2 = (const float*)d_in[4];
  const float* W3 = (const float*)d_in[5];
  const float* b3 = (const float*)d_in[6];
  const float* Wy = (const float*)d_in[7];
  float* out = (float*)d_out;

  // workspace layout: [W1h: 65536 f16][W2h: 262144 f16] = 640 KB
  _Float16* W1hp = (_Float16*)d_ws;
  _Float16* W2hp = W1hp + W1_ELEMS;

  // one-time weight down-conversion (L2-resident afterwards)
  cvt_weights_kernel<<<(W2_ELEMS + 255) / 256, 256, 0, stream>>>(W1, W2, W1hp,
                                                                 W2hp);

  const int Bn = in_sizes[0] / SS;      // 131072
  dim3 grid(Bn / BT, PP);
  puf_fused_kernel<<<grid, 256, 0, stream>>>(x, W1hp, b1, W2hp, b2, W3, b3, Wy,
                                             out, Bn);
}
